// Attention_27943057228032
// MI455X (gfx1250) — matrix operand order repaired and checked by reading, not compiled, not run on hardware
//
#include <hip/hip_runtime.h>
#include <hip/hip_bf16.h>
#include <math.h>

// Problem constants (reference: B=2, T=2048, D=1024, H=16, hd=64)
#define BSZ 2
#define TSEQ 2048
#define DMODEL 1024
#define NHEAD 16
#define HDIM 64
#define EPSN 1e-4f

typedef __attribute__((ext_vector_type(16))) _Float16 v16h;
typedef __attribute__((ext_vector_type(8)))  _Float16 v8h;
typedef __attribute__((ext_vector_type(8)))  float    v8f;

// ---------------------------------------------------------------------------
// WMMA fragment loaders (wave32, v_wmma_f32_16x16x32_f16 layouts per CDNA5 ISA)
// A (16x32 f16, MxK): lane L<16 -> row L,   K = {0..7} in halves 0..7, {16..23} in 8..15
//                     lane L>=16 -> row L-16, K = {8..15} and {24..31}
// ---------------------------------------------------------------------------
__device__ __forceinline__ v16h load_a_frag(const _Float16* __restrict__ base,
                                            int ld, int lane) {
  const int row = lane & 15;
  const int ko  = (lane < 16) ? 0 : 8;
  const _Float16* p = base + (size_t)row * ld + ko;
  v8h lo = *reinterpret_cast<const v8h*>(p);        // K = ko .. ko+7
  v8h hi = *reinterpret_cast<const v8h*>(p + 16);   // K = 16+ko .. 16+ko+7
  v16h a;
#pragma unroll
  for (int i = 0; i < 8; ++i) { a[i] = lo[i]; a[8 + i] = hi[i]; }
  return a;
}

// B (32x16 f16, KxN): lane holds one column n = lane&15; lanes 0-15 K={0..7,16..23},
// lanes 16-31 K={8..15,24..31}. Storage is B^T row-major: element(k,n) = base[n*ld + k].
__device__ __forceinline__ v16h load_b_frag(const _Float16* __restrict__ base,
                                            int ld, int lane) {
  const int col = lane & 15;
  const int ko  = (lane < 16) ? 0 : 8;
  const _Float16* p = base + (size_t)col * ld + ko;
  v8h lo = *reinterpret_cast<const v8h*>(p);        // K = ko .. ko+7
  v8h hi = *reinterpret_cast<const v8h*>(p + 16);   // K = 16+ko .. 16+ko+7
  v16h b;
#pragma unroll
  for (int i = 0; i < 8; ++i) { b[i] = lo[i]; b[8 + i] = hi[i]; }
  return b;
}

__device__ __forceinline__ float redmax16(float v) {
#pragma unroll
  for (int m = 8; m >= 1; m >>= 1) v = fmaxf(v, __shfl_xor(v, m, 32));
  return v;
}

// ---------------------------------------------------------------------------
// Kernel 1: magnitude-preserving weight normalization + f16 cast (wave/row)
// ---------------------------------------------------------------------------
__global__ void wnorm_cast_kernel(const float* __restrict__ W,
                                  _Float16* __restrict__ Wh,
                                  int rows, int K) {
  const int wave = (int)((blockIdx.x * blockDim.x + threadIdx.x) >> 5);
  const int lane = threadIdx.x & 31;
  if (wave >= rows) return;
  const float* w = W + (size_t)wave * K;
  float ss = 0.f;
  for (int k = lane; k < K; k += 32) { float v = w[k]; ss += v * v; }
#pragma unroll
  for (int m = 16; m >= 1; m >>= 1) ss += __shfl_xor(ss, m, 32);
  const float rsqK  = rsqrtf((float)K);
  const float scale = rsqK / (EPSN + sqrtf(ss) * rsqK);
  _Float16* o = Wh + (size_t)wave * K;
  for (int k = lane; k < K; k += 32) o[k] = (_Float16)(w[k] * scale);
}

// Kernel 2: f32 -> f16 cast of activations
__global__ void cast_f32_f16_kernel(const float* __restrict__ in,
                                    _Float16* __restrict__ out, int n) {
  int i = blockIdx.x * blockDim.x + threadIdx.x;
  if (i < n) out[i] = (_Float16)in[i];
}

// ---------------------------------------------------------------------------
// Kernel 3/6: GEMM  C[M,N] = A[M,K] * Bt[N,K]^T   (f16 in, f32 out)
// One wave computes a 32x64 tile (8 WMMAs / 12 b128 loads per 32-k-step).
// Ping-pong pipelining: the K loop is unrolled by two 32-steps alternating two
// STATICALLY NAMED fragment buffers, so there are no rotation copies and no
// conditional loads (tail peeled; K % 64 == 0 at all call sites). Live set:
// 64 acc + 48 + 48 frag VGPRs -> no spills, loads overlap WMMAs.
// ---------------------------------------------------------------------------
__global__ __launch_bounds__(256)
void gemm_f16_wmma_kernel(const _Float16* __restrict__ A,
                          const _Float16* __restrict__ Bt,
                          float* __restrict__ C,
                          int M, int N, int K) {
  const int wave = (int)((blockIdx.x * blockDim.x + threadIdx.x) >> 5);
  const int lane = threadIdx.x & 31;
  const int ntiles = N >> 6;                 // tiles of 64 output columns
  const int total  = (M >> 5) * ntiles;      // 32-row M tiles
  if (wave >= total) return;
  const int mtile = wave / ntiles;
  const int nt    = wave % ntiles;

  const _Float16* A0 = A + (size_t)(mtile * 32) * K;
  const _Float16* A1 = A0 + (size_t)16 * K;
  const _Float16* B0 = Bt + (size_t)(nt * 64) * K;

  v8f acc[8];
#pragma unroll
  for (int i = 0; i < 8; ++i) acc[i] = v8f{};

  v16h a0A, a1A, bA0, bA1, bA2, bA3;   // ping buffer
  v16h a0B, a1B, bB0, bB1, bB2, bB3;   // pong buffer

#define LOADSTEP(kk, a0, a1, b0, b1, b2, b3)                                   \
  do {                                                                         \
    a0 = load_a_frag(A0 + (kk), K, lane);                                      \
    a1 = load_a_frag(A1 + (kk), K, lane);                                      \
    b0 = load_b_frag(B0 + (size_t)0  * K + (kk), K, lane);                     \
    b1 = load_b_frag(B0 + (size_t)16 * K + (kk), K, lane);                     \
    b2 = load_b_frag(B0 + (size_t)32 * K + (kk), K, lane);                     \
    b3 = load_b_frag(B0 + (size_t)48 * K + (kk), K, lane);                     \
  } while (0)

#define COMPSTEP(a0, a1, b0, b1, b2, b3)                                       \
  do {                                                                         \
    acc[0] = __builtin_amdgcn_wmma_f32_16x16x32_f16(false, a0, false, b0, (short)0, acc[0], false, false); \
    acc[4] = __builtin_amdgcn_wmma_f32_16x16x32_f16(false, a1, false, b0, (short)0, acc[4], false, false); \
    acc[1] = __builtin_amdgcn_wmma_f32_16x16x32_f16(false, a0, false, b1, (short)0, acc[1], false, false); \
    acc[5] = __builtin_amdgcn_wmma_f32_16x16x32_f16(false, a1, false, b1, (short)0, acc[5], false, false); \
    acc[2] = __builtin_amdgcn_wmma_f32_16x16x32_f16(false, a0, false, b2, (short)0, acc[2], false, false); \
    acc[6] = __builtin_amdgcn_wmma_f32_16x16x32_f16(false, a1, false, b2, (short)0, acc[6], false, false); \
    acc[3] = __builtin_amdgcn_wmma_f32_16x16x32_f16(false, a0, false, b3, (short)0, acc[3], false, false); \
    acc[7] = __builtin_amdgcn_wmma_f32_16x16x32_f16(false, a1, false, b3, (short)0, acc[7], false, false); \
  } while (0)

  LOADSTEP(0, a0A, a1A, bA0, bA1, bA2, bA3);
  int k0 = 0;
  for (; k0 + 64 < K; k0 += 64) {
    LOADSTEP(k0 + 32, a0B, a1B, bB0, bB1, bB2, bB3);
    COMPSTEP(a0A, a1A, bA0, bA1, bA2, bA3);
    LOADSTEP(k0 + 64, a0A, a1A, bA0, bA1, bA2, bA3);
    COMPSTEP(a0B, a1B, bB0, bB1, bB2, bB3);
  }
  // tail: k0 == K-64; compute last two 32-steps
  LOADSTEP(k0 + 32, a0B, a1B, bB0, bB1, bB2, bB3);
  COMPSTEP(a0A, a1A, bA0, bA1, bA2, bA3);
  COMPSTEP(a0B, a1B, bB0, bB1, bB2, bB3);
#undef LOADSTEP
#undef COMPSTEP

  const int row0 = (lane < 16) ? 0 : 8;
  const int col  = lane & 15;
#pragma unroll
  for (int i = 0; i < 2; ++i)
#pragma unroll
    for (int j = 0; j < 4; ++j)
#pragma unroll
      for (int r = 0; r < 8; ++r)
        C[(size_t)(mtile * 32 + i * 16 + row0 + r) * N + nt * 64 + j * 16 + col] =
            acc[i * 4 + j][r];
}

// ---------------------------------------------------------------------------
// Kernel 4: qkv post-process. mp-normalize q,k over head dim; cast to f16.
// Q,K stored (b*H+h, T, 64) row-major; V stored transposed (b*H+h, 64, T).
// ---------------------------------------------------------------------------
__global__ void qkv_post_kernel(const float* __restrict__ qkv,
                                _Float16* __restrict__ Qh,
                                _Float16* __restrict__ Kh,
                                _Float16* __restrict__ Vt) {
  int idx = blockIdx.x * blockDim.x + threadIdx.x;     // over B*H*T
  if (idx >= BSZ * NHEAD * TSEQ) return;
  const int b = idx / (NHEAD * TSEQ);
  const int r = idx % (NHEAD * TSEQ);
  const int h = r / TSEQ;
  const int t = r % TSEQ;
  const float* base = qkv + ((size_t)b * TSEQ + t) * (3 * DMODEL);
  const float* q = base + h * HDIM;
  const float* k = base + DMODEL + h * HDIM;
  const float* v = base + 2 * DMODEL + h * HDIM;
  float sq = 0.f, sk = 0.f;
#pragma unroll 8
  for (int d = 0; d < HDIM; ++d) { sq += q[d] * q[d]; sk += k[d] * k[d]; }
  const float qs = 1.f / (EPSN + sqrtf(sq) * 0.125f);  // 1/sqrt(64) = 0.125
  const float ks = 1.f / (EPSN + sqrtf(sk) * 0.125f);
  const size_t obase = ((size_t)(b * NHEAD + h) * TSEQ + t) * HDIM;
  const size_t vbase = (size_t)(b * NHEAD + h) * HDIM * TSEQ + t;
#pragma unroll 8
  for (int d = 0; d < HDIM; ++d) {
    Qh[obase + d] = (_Float16)(q[d] * qs);
    Kh[obase + d] = (_Float16)(k[d] * ks);
    Vt[vbase + (size_t)d * TSEQ] = (_Float16)v[d];
  }
}

// ---------------------------------------------------------------------------
// Kernel 5: flash attention. One wave per (b,h, 16-query tile).
// Keys in tiles of 32: 4 score WMMAs, online softmax (row-max via 4-shfl
// reduction; row-SUM via a 9th WMMA against an all-ones B matrix), P staged
// through LDS (C-layout -> A-fragment transpose, per-wave s_wait_dscnt fence),
// 4 P*V WMMAs. Next K/V tiles prefetched with global_prefetch_b8.
// ---------------------------------------------------------------------------
#define ATT_WAVES 8
__global__ __launch_bounds__(256)
void attention_kernel(const _Float16* __restrict__ Qh,
                      const _Float16* __restrict__ Kh,
                      const _Float16* __restrict__ Vt,
                      _Float16* __restrict__ Oh) {
  __shared__ _Float16 Pbuf[ATT_WAVES][16][32];

  const int wave  = (int)((blockIdx.x * blockDim.x + threadIdx.x) >> 5);
  const int wib   = (threadIdx.x >> 5);
  const int lane  = threadIdx.x & 31;
  const int qtpbh = TSEQ / 16;
  if (wave >= BSZ * NHEAD * qtpbh) return;
  const int bh = wave / qtpbh;
  const int qt = wave % qtpbh;
  const int b  = bh / NHEAD;
  const int h  = bh % NHEAD;

  const _Float16* Qbase = Qh + ((size_t)bh * TSEQ + qt * 16) * HDIM;
  const _Float16* Kbh   = Kh + (size_t)bh * TSEQ * HDIM;
  const _Float16* Vbh   = Vt + (size_t)bh * HDIM * TSEQ;

  // Q fragments for K-dim 0..31 and 32..63 (loaded once, reused 64x)
  const v16h aq0 = load_a_frag(Qbase, HDIM, lane);
  const v16h aq1 = load_a_frag(Qbase + 32, HDIM, lane);

  v16h ones;
#pragma unroll
  for (int i = 0; i < 16; ++i) ones[i] = (_Float16)1.0f;

  v8f o[4] = {v8f{}, v8f{}, v8f{}, v8f{}};
  v8f lacc = v8f{};                 // row-sum accumulator (every col = rowsum)
  float mrow[8];
#pragma unroll
  for (int r = 0; r < 8; ++r) mrow[r] = -3.0e38f;

  const float sc = 0.125f;  // 1/sqrt(hd)

  for (int kt = 0; kt < TSEQ; kt += 32) {
    // prefetch next key/value tiles while this iteration computes
    if (kt + 32 < TSEQ) {
      const int col = lane & 15;
      const int kof = (lane < 16) ? 0 : 16;
      __builtin_prefetch(Kbh + (size_t)(kt + 32 + col) * HDIM, 0, 3);
      __builtin_prefetch(Kbh + (size_t)(kt + 48 + col) * HDIM, 0, 3);
      __builtin_prefetch(Vbh + (size_t)col * TSEQ + kt + 32 + kof, 0, 3);
      __builtin_prefetch(Vbh + (size_t)(32 + col) * TSEQ + kt + 32 + kof, 0, 3);
    }

    // scores for 32 keys: two 16x16 C tiles, each K-dim 64 (two chained WMMAs)
    v8f s0 = v8f{}, s1 = v8f{};
    {
      const _Float16* kb = Kbh + (size_t)kt * HDIM;
      v16h b0 = load_b_frag(kb, HDIM, lane);
      s0 = __builtin_amdgcn_wmma_f32_16x16x32_f16(false, aq0, false, b0, (short)0, s0, false, false);
      v16h b1 = load_b_frag(kb + 32, HDIM, lane);
      s0 = __builtin_amdgcn_wmma_f32_16x16x32_f16(false, aq1, false, b1, (short)0, s0, false, false);
      const _Float16* kb2 = Kbh + (size_t)(kt + 16) * HDIM;
      v16h b2 = load_b_frag(kb2, HDIM, lane);
      s1 = __builtin_amdgcn_wmma_f32_16x16x32_f16(false, aq0, false, b2, (short)0, s1, false, false);
      v16h b3 = load_b_frag(kb2 + 32, HDIM, lane);
      s1 = __builtin_amdgcn_wmma_f32_16x16x32_f16(false, aq1, false, b3, (short)0, s1, false, false);
    }

    // online softmax update (rows live across 16-lane halves)
#pragma unroll
    for (int r = 0; r < 8; ++r) {
      const float a0 = s0[r] * sc;
      const float a1 = s1[r] * sc;
      const float rm = redmax16(fmaxf(a0, a1));
      const float mn = fmaxf(mrow[r], rm);
      const float alpha = __expf(mrow[r] - mn);
      const float p0 = __expf(a0 - mn);
      const float p1 = __expf(a1 - mn);
      mrow[r] = mn;
      lacc[r] *= alpha;
#pragma unroll
      for (int j = 0; j < 4; ++j) o[j][r] *= alpha;
      // stage P into LDS (row-major 16x32) for the A-fragment reload
      const int row = r + ((lane < 16) ? 0 : 8);
      const int col = lane & 15;
      Pbuf[wib][row][col]      = (_Float16)p0;
      Pbuf[wib][row][col + 16] = (_Float16)p1;
    }
    asm volatile("s_wait_dscnt 0" ::: "memory");  // per-wave LDS RAW fence

    // P(16x32) A-fragment from LDS; row-sum via WMMA against ones
    const v16h ap = load_a_frag(&Pbuf[wib][0][0], 32, lane);
    lacc = __builtin_amdgcn_wmma_f32_16x16x32_f16(false, ap, false, ones, (short)0, lacc, false, false);
#pragma unroll
    for (int j = 0; j < 4; ++j) {
      const _Float16* vb = Vbh + (size_t)(j * 16) * TSEQ + kt;
      v16h bv = load_b_frag(vb, TSEQ, lane);
      o[j] = __builtin_amdgcn_wmma_f32_16x16x32_f16(false, ap, false, bv, (short)0, o[j], false, false);
    }
  }

  // normalize and write f16 in (B, T, D) layout for the output projection
  const int row0 = (lane < 16) ? 0 : 8;
  const int col  = lane & 15;
#pragma unroll
  for (int r = 0; r < 8; ++r) {
    const float inv = 1.f / lacc[r];
    const int t = qt * 16 + row0 + r;
#pragma unroll
    for (int j = 0; j < 4; ++j) {
      Oh[((size_t)b * TSEQ + t) * DMODEL + h * HDIM + j * 16 + col] =
          (_Float16)(o[j][r] * inv);
    }
  }
}

// ---------------------------------------------------------------------------
// Workspace layout (bytes)
// ---------------------------------------------------------------------------
#define OFF_WQKVH  ((size_t)0)                       // 3072*1024 f16 = 6 MB
#define OFF_WOUTH  ((size_t)(6u  * 1024 * 1024))     // 1024*1024 f16 = 2 MB
#define OFF_XH     ((size_t)(8u  * 1024 * 1024))     // 4096*1024 f16 = 8 MB
#define OFF_QKV32  ((size_t)(16u * 1024 * 1024))     // 4096*3072 f32 = 48 MB
#define OFF_QH     ((size_t)(64u * 1024 * 1024))     // 8 MB
#define OFF_KH     ((size_t)(72u * 1024 * 1024))     // 8 MB
#define OFF_VT     ((size_t)(80u * 1024 * 1024))     // 8 MB
#define OFF_AOUT   ((size_t)(88u * 1024 * 1024))     // 8 MB   (total 96 MB)

extern "C" void kernel_launch(void* const* d_in, const int* in_sizes, int n_in,
                              void* d_out, int out_size, void* d_ws, size_t ws_size,
                              hipStream_t stream) {
  const float* x     = (const float*)d_in[0];   // (2, 2048, 1024)
  const float* w_qkv = (const float*)d_in[1];   // (3072, 1024)
  const float* w_out = (const float*)d_in[2];   // (1024, 1024)
  float* out = (float*)d_out;                   // (2, 2048, 1024)

  char* ws = (char*)d_ws;
  _Float16* wqkv_h = (_Float16*)(ws + OFF_WQKVH);
  _Float16* wout_h = (_Float16*)(ws + OFF_WOUTH);
  _Float16* xh     = (_Float16*)(ws + OFF_XH);
  float*    qkv32  = (float*)   (ws + OFF_QKV32);
  _Float16* Qhp    = (_Float16*)(ws + OFF_QH);
  _Float16* Khp    = (_Float16*)(ws + OFF_KH);
  _Float16* Vtp    = (_Float16*)(ws + OFF_VT);
  _Float16* aout   = (_Float16*)(ws + OFF_AOUT);

  const int M = BSZ * TSEQ;          // 4096
  const int K = DMODEL;              // 1024
  const int Nqkv = 3 * DMODEL;       // 3072

  // 1) normalize + cast weights (one wave per row)
  wnorm_cast_kernel<<<(Nqkv * 32 + 255) / 256, 256, 0, stream>>>(w_qkv, wqkv_h, Nqkv, K);
  wnorm_cast_kernel<<<(DMODEL * 32 + 255) / 256, 256, 0, stream>>>(w_out, wout_h, DMODEL, K);

  // 2) cast x -> f16
  {
    int n = M * K;
    cast_f32_f16_kernel<<<(n + 255) / 256, 256, 0, stream>>>(x, xh, n);
  }

  // 3) qkv GEMM: (4096 x 3072) = xh (4096 x 1024) * wqkv_h^T
  {
    int waves = (M / 32) * (Nqkv / 64);          // 6144
    gemm_f16_wmma_kernel<<<(waves * 32 + 255) / 256, 256, 0, stream>>>(
        xh, wqkv_h, qkv32, M, Nqkv, K);
  }

  // 4) split heads, mp-normalize q/k, cast, transpose V
  {
    int n = BSZ * NHEAD * TSEQ;                  // 65536
    qkv_post_kernel<<<(n + 255) / 256, 256, 0, stream>>>(qkv32, Qhp, Khp, Vtp);
  }

  // 5) flash attention (one wave per 16-query tile)
  {
    int waves = BSZ * NHEAD * (TSEQ / 16);       // 4096
    attention_kernel<<<(waves * 32 + 255) / 256, 256, 0, stream>>>(Qhp, Khp, Vtp, aout);
  }

  // 6) output projection: (4096 x 1024) = aout * wout_h^T  -> f32 d_out
  {
    int waves = (M / 32) * (DMODEL / 64);        // 2048
    gemm_f16_wmma_kernel<<<(waves * 32 + 255) / 256, 256, 0, stream>>>(
        aout, wout_h, out, M, DMODEL, K);
  }
}
